// CubeRE_19877108646156
// MI455X (gfx1250) — compile-verified, matrix operand order checked by reading, not run on hardware
//
#include <hip/hip_runtime.h>
#include <math.h>

// ---------------------------------------------------------------------------
// CubeRE fused forward for MI455X (gfx1250, wave32, WMMA bf16 path).
// Shapes: B=4 S=192 H=M=768 V=66 O=20 K=64.
// Heavy GEMMs run on v_wmma_f32_16x16x32_bf16 (fp32 accumulate).
// ---------------------------------------------------------------------------

typedef __attribute__((ext_vector_type(16))) __bf16 v16bf;
typedef __attribute__((ext_vector_type(8)))  float  v8f;

#define CB 4
#define CS 192
#define CH 768
#define CM 768
#define CV 66
#define CO 20
#define CK 64

__device__ inline float gelu_exact(float x) {
  return 0.5f * x * (1.0f + erff(x * 0.70710678118654752440f));
}

__device__ inline v8f zero8() {
  v8f z;
#pragma unroll
  for (int i = 0; i < 8; ++i) z[i] = 0.0f;
  return z;
}

// Load one 16x32 bf16 fragment (A-matrix layout, ISA 7.12.2; Bt rows use the
// same layout since B^T(n,k) mirrors A(r,k)).  base: row-major [rows][ldk].
__device__ inline v16bf load_frag(const __bf16* base, int row, int ldk, int kc, int lane) {
  int r  = row + (lane & 15);
  int hi = (lane >> 4) & 1;
  const unsigned* q = (const unsigned*)((const unsigned short*)base +
                                        (size_t)r * ldk + kc + (hi << 3));
  union { v16bf v; unsigned u[8]; } f;
  f.u[0] = q[0];   // K 0-1   (+8 if hi)
  f.u[1] = q[1];   // K 2-3
  f.u[2] = q[2];   // K 4-5
  f.u[3] = q[3];   // K 6-7
  f.u[4] = q[8];   // K 16-17
  f.u[5] = q[9];   // K 18-19
  f.u[6] = q[10];  // K 20-21
  f.u[7] = q[11];  // K 22-23
  return f.v;
}

// ---------------------------------------------------------------------------
// Elementwise helpers
// ---------------------------------------------------------------------------
__global__ void k_cvt_bf16(const float* __restrict__ src, __bf16* __restrict__ dst, int n) {
  int i = blockIdx.x * blockDim.x + threadIdx.x;
  if (i < n) dst[i] = (__bf16)src[i];
}

// dst[n*K + k] = (n < ncap) ? src[(rowoff+k)*ldsrc + n] : 0   (build B^T in bf16)
__global__ void k_transpose_cvt(const float* __restrict__ src, __bf16* __restrict__ dst,
                                int Kd, int Nd, int ldsrc, int rowoff, int ncap) {
  int i = blockIdx.x * blockDim.x + threadIdx.x;
  if (i >= Nd * Kd) return;
  int n = i / Kd, k = i - n * Kd;
  float v = (n < ncap) ? src[(size_t)(rowoff + k) * ldsrc + n] : 0.0f;
  dst[(size_t)n * Kd + k] = (__bf16)v;
}

__global__ void k_zero4(float* p) {
  int i = threadIdx.x;
  if (i < 4) p[i] = 0.0f;
}

// ---------------------------------------------------------------------------
// Generic bf16 WMMA GEMM: D[m,n] = sum_k A[m,k]*Bt[n,k], fp32 out.
// One wave computes a 16x64 tile (4 accumulators). All dims multiples of 16/64.
// ---------------------------------------------------------------------------
__global__ void k_gemm_bf16(const __bf16* __restrict__ A, const __bf16* __restrict__ Bt,
                            float* __restrict__ C, int Kd, int ldc,
                            int nTilesM, int nTilesN64, int nBatch,
                            long long strideA, long long strideB, long long strideC) {
  int lane = threadIdx.x & 31;
  int wid  = blockIdx.x * (blockDim.x >> 5) + (threadIdx.x >> 5);
  int per  = nTilesM * nTilesN64;
  if (wid >= per * nBatch) return;
  int bb = wid / per, r = wid - bb * per;
  int mt = r / nTilesN64, ng = r - mt * nTilesN64;
  const __bf16* Ab = A  + (long long)bb * strideA;
  const __bf16* Bb = Bt + (long long)bb * strideB;
  float*        Cb = C  + (long long)bb * strideC;
  int m0 = mt * 16, n0 = ng * 64;

  v8f acc0 = zero8(), acc1 = zero8(), acc2 = zero8(), acc3 = zero8();
  for (int kc = 0; kc < Kd; kc += 32) {
    v16bf a  = load_frag(Ab, m0,      Kd, kc, lane);
    v16bf b0 = load_frag(Bb, n0,      Kd, kc, lane);
    v16bf b1 = load_frag(Bb, n0 + 16, Kd, kc, lane);
    v16bf b2 = load_frag(Bb, n0 + 32, Kd, kc, lane);
    v16bf b3 = load_frag(Bb, n0 + 48, Kd, kc, lane);
    acc0 = __builtin_amdgcn_wmma_f32_16x16x32_bf16(false, a, false, b0, (short)0, acc0, false, false);
    acc1 = __builtin_amdgcn_wmma_f32_16x16x32_bf16(false, a, false, b1, (short)0, acc1, false, false);
    acc2 = __builtin_amdgcn_wmma_f32_16x16x32_bf16(false, a, false, b2, (short)0, acc2, false, false);
    acc3 = __builtin_amdgcn_wmma_f32_16x16x32_bf16(false, a, false, b3, (short)0, acc3, false, false);
  }
  int n = lane & 15, hi = (lane >> 4) & 1;
#pragma unroll
  for (int v = 0; v < 8; ++v) {
    int m = v + (hi << 3);
    float* row = Cb + (long long)(m0 + m) * ldc + n0 + n;
    row[0]  = acc0[v];
    row[16] = acc1[v];
    row[32] = acc2[v];
    row[48] = acc3[v];
  }
}

// ---------------------------------------------------------------------------
// Fused joint_score: pair = gelu(hh[b,i,:] + ht[b,j,:] + b1), score = pair@Wf + bf.
// One wave per (b, i, 16-j block); A tile built in LDS each K-step; 5 N tiles.
// ---------------------------------------------------------------------------
__global__ void k_joint(const float* __restrict__ hh, const float* __restrict__ ht,
                        const float* __restrict__ b1, const __bf16* __restrict__ WfT,
                        const float* __restrict__ bfv, float* __restrict__ out) {
  int jb = blockIdx.x;      // 0..11
  int i  = blockIdx.y;      // 0..191
  int b  = blockIdx.z;      // 0..3
  int lane = threadIdx.x & 31;

  __shared__ __bf16 atile[16 * 32];
  __shared__ float  hb[32];

  const float* hhrow  = hh + ((long long)(b * CS + i)) * CM;
  const float* htbase = ht + ((long long)(b * CS + jb * 16)) * CM;

  v8f acc[5];
#pragma unroll
  for (int t = 0; t < 5; ++t) acc[t] = zero8();

  for (int kc = 0; kc < CM; kc += 32) {
    hb[lane] = hhrow[kc + lane] + b1[kc + lane];
    __syncthreads();
#pragma unroll
    for (int j = 0; j < 16; ++j) {
      float v = hb[lane] + htbase[(long long)j * CM + kc + lane];
      atile[j * 32 + lane] = (__bf16)gelu_exact(v);
    }
    __syncthreads();
    v16bf a = load_frag(atile, 0, 32, 0, lane);
#pragma unroll
    for (int t = 0; t < 5; ++t) {
      v16bf bt = load_frag(WfT, t * 16, CM, kc, lane);
      acc[t] = __builtin_amdgcn_wmma_f32_16x16x32_bf16(false, a, false, bt, (short)0, acc[t], false, false);
    }
    __syncthreads();  // protect atile/hb against next-iteration overwrite
  }

  int ncol = lane & 15, hi = (lane >> 4) & 1;
#pragma unroll
  for (int t = 0; t < 5; ++t) {
    int n = t * 16 + ncol;
    if (n < CV) {
      float bias = bfv[n];
#pragma unroll
      for (int v = 0; v < 8; ++v) {
        int m = v + (hi << 3);
        int j = jb * 16 + m;
        out[(((long long)(b * CS + i)) * CS + j) * CV + n] = acc[t][v] + bias;
      }
    }
  }
}

// ---------------------------------------------------------------------------
// Top-k (K=64) over diagonal entity scores; lax.top_k tie-break (lowest index).
// ---------------------------------------------------------------------------
__global__ void k_topk(const float* __restrict__ joint, const unsigned char* __restrict__ jmask,
                       int* __restrict__ indices) {
  int b = blockIdx.x;
  int t = threadIdx.x;  // 256
  __shared__ float sc[256];
  __shared__ float rv[256];
  __shared__ int   ri[256];

  float val = -3.4e38f;
  if (t < CS) {
    const float* row = joint + ((((long long)(b * CS + t)) * CS + t)) * CV;
    float mx = row[1];
#pragma unroll
    for (int v = 2; v <= 10; ++v) mx = fmaxf(mx, row[v]);  // ENT_LABEL = 1..10
    val = mx;
  }
  sc[t] = val;
  rv[t] = (t < CS) ? val : 3.4e38f;
  __syncthreads();
  for (int off = 128; off > 0; off >>= 1) {
    if (t < off) rv[t] = fminf(rv[t], rv[t + off]);
    __syncthreads();
  }
  float vmin = rv[0];
  __syncthreads();
  if (t < CS) {
    unsigned char m = jmask[((long long)(b * CS + t)) * CS + t];
    if (!m) sc[t] = vmin;
  }
  __syncthreads();

  for (int k = 0; k < CK; ++k) {
    rv[t] = (t < CS) ? sc[t] : -3.4e38f;
    ri[t] = t;
    __syncthreads();
    for (int off = 128; off > 0; off >>= 1) {
      if (t < off) {
        if (rv[t + off] > rv[t] || (rv[t + off] == rv[t] && ri[t + off] < ri[t])) {
          rv[t] = rv[t + off];
          ri[t] = ri[t + off];
        }
      }
      __syncthreads();
    }
    if (t == 0) {
      indices[b * CK + k] = ri[0];
      sc[ri[0]] = -3.4e38f;
    }
    __syncthreads();
  }
}

// pruned gather: fp32 + bf16 copies
__global__ void k_prune(const float* __restrict__ x, const int* __restrict__ idx,
                        float* __restrict__ pf, __bf16* __restrict__ pb) {
  int i = blockIdx.x * blockDim.x + threadIdx.x;  // B*K*H
  if (i >= CB * CK * CH) return;
  int h = i % CH;
  int r = i / CH;
  int k = r & 63, b = r >> 6;
  int ix = idx[b * CK + k];
  float v = x[((long long)(b * CS + ix)) * CH + h];
  pf[i] = v;
  pb[i] = (__bf16)v;
}

// pair2 = gelu(p_head[b,x,:] + p_tail[b,y,:] + b2) -> bf16 [B,K,K,M]
__global__ void k_pair2(const float* __restrict__ ph, const float* __restrict__ pt,
                        const float* __restrict__ b2, __bf16* __restrict__ out) {
  int i = blockIdx.x * blockDim.x + threadIdx.x;  // B*K*K*M
  if (i >= CB * CK * CK * CM) return;
  int m = i % CM;
  int r = i / CM;
  int y = r & 63;  r >>= 6;
  int x = r & 63;  int b = r >> 6;
  float v = ph[((long long)(b * CK + x)) * CM + m] + pt[((long long)(b * CK + y)) * CM + m] + b2[m];
  out[i] = (__bf16)gelu_exact(v);
}

// t[(b,z),(o,i)] fp32 -> tT_bf[b][(z*O+o)][i] bf16  (B^T for the q_score GEMM)
__global__ void k_tT(const float* __restrict__ tf, __bf16* __restrict__ tT) {
  int i = blockIdx.x * blockDim.x + threadIdx.x;  // B*K*O*M
  if (i >= CB * CK * CO * CM) return;
  int m = i % CM;
  int r = i / CM;
  int o = r % CO;  r /= CO;
  int z = r & 63;  int b = r >> 6;
  float v = tf[((long long)(b * CK + z)) * (CO * CM) + (long long)o * CM + m];
  tT[i] = (__bf16)v;  // i decomposes exactly as ((b*K+z)*O+o)*M+m
}

// ---------------------------------------------------------------------------
// Masked cross-entropy reductions
// ---------------------------------------------------------------------------
__global__ void k_ce_joint(const float* __restrict__ joint, const int* __restrict__ lab,
                           const unsigned char* __restrict__ msk, float* __restrict__ partials) {
  int t = blockIdx.x * blockDim.x + threadIdx.x;
  const int total = CB * CS * CS;
  float nll = 0.0f, m = 0.0f;
  if (t < total) {
    const float* lg = joint + (long long)t * CV;
    float mx = lg[0];
    for (int v = 1; v < CV; ++v) mx = fmaxf(mx, lg[v]);
    float s = 0.0f;
    for (int v = 0; v < CV; ++v) s += expf(lg[v] - mx);
    int l = lab[t];
    nll = mx + logf(s) - lg[l];
    m = msk[t] ? 1.0f : 0.0f;
  }
  __shared__ float s1[256], s2[256];
  s1[threadIdx.x] = nll * m;
  s2[threadIdx.x] = m;
  __syncthreads();
  for (int off = 128; off > 0; off >>= 1) {
    if (threadIdx.x < off) {
      s1[threadIdx.x] += s1[threadIdx.x + off];
      s2[threadIdx.x] += s2[threadIdx.x + off];
    }
    __syncthreads();
  }
  if (threadIdx.x == 0) {
    atomicAdd(&partials[0], s1[0]);
    atomicAdd(&partials[1], s2[0]);
  }
}

__global__ void k_ce_q(const float* __restrict__ q, const int* __restrict__ qlabS,
                       const unsigned char* __restrict__ qmskS, const int* __restrict__ idx,
                       float* __restrict__ partials) {
  int t = blockIdx.x * blockDim.x + threadIdx.x;  // B*K*K*K
  const int total = CB * CK * CK * CK;
  float nll = 0.0f, m = 0.0f;
  if (t < total) {
    int z = t & 63, y = (t >> 6) & 63, x = (t >> 12) & 63, b = t >> 18;
    int ix = idx[b * CK + x], iy = idx[b * CK + y], iz = idx[b * CK + z];
    long long g = (((long long)(b * CS + ix)) * CS + iy) * CS + iz;
    const float* lg = q + (long long)t * CO;
    float mx = lg[0];
    for (int v = 1; v < CO; ++v) mx = fmaxf(mx, lg[v]);
    float s = 0.0f;
    for (int v = 0; v < CO; ++v) s += expf(lg[v] - mx);
    int l = qlabS[g];
    nll = mx + logf(s) - lg[l];
    m = qmskS[g] ? 1.0f : 0.0f;
  }
  __shared__ float s1[256], s2[256];
  s1[threadIdx.x] = nll * m;
  s2[threadIdx.x] = m;
  __syncthreads();
  for (int off = 128; off > 0; off >>= 1) {
    if (threadIdx.x < off) {
      s1[threadIdx.x] += s1[threadIdx.x + off];
      s2[threadIdx.x] += s2[threadIdx.x + off];
    }
    __syncthreads();
  }
  if (threadIdx.x == 0) {
    atomicAdd(&partials[2], s1[0]);
    atomicAdd(&partials[3], s2[0]);
  }
}

__global__ void k_final(const float* __restrict__ p, float* __restrict__ out) {
  out[0] = p[0] / fmaxf(p[1], 1.0f) + p[2] / fmaxf(p[3], 1.0f);
}

// ---------------------------------------------------------------------------
// Host orchestration
// ---------------------------------------------------------------------------
extern "C" void kernel_launch(void* const* d_in, const int* in_sizes, int n_in,
                              void* d_out, int out_size, void* d_ws, size_t ws_size,
                              hipStream_t stream) {
  (void)in_sizes; (void)n_in; (void)out_size; (void)ws_size;
  const float*         x     = (const float*)d_in[0];
  const unsigned char* jmask = (const unsigned char*)d_in[1];
  const int*           jlab  = (const int*)d_in[2];
  const unsigned char* qmask = (const unsigned char*)d_in[3];
  const int*           qlab  = (const int*)d_in[4];
  // d_in[5] = topk scalar (compile-time K=64)
  const float* W1  = (const float*)d_in[6];
  const float* b1  = (const float*)d_in[7];
  const float* Wf  = (const float*)d_in[8];
  const float* bfv = (const float*)d_in[9];
  const float* W2  = (const float*)d_in[10];
  const float* b2  = (const float*)d_in[11];
  const float* U   = (const float*)d_in[12];

  float* out   = (float*)d_out;
  float* joint = out + 1;
  float* qsc   = out + 1 + (long long)CB * CS * CS * CV;

  // workspace carve-up
  char*  w   = (char*)d_ws;
  size_t off = 0;
  auto alloc = [&](size_t bytes) -> char* {
    char* p = w + off;
    off += (bytes + 255) & ~size_t(255);
    return p;
  };
  __bf16* x_bf   = (__bf16*)alloc((size_t)CB * CS * CH * 2);
  __bf16* W1aT   = (__bf16*)alloc((size_t)CM * CH * 2);
  __bf16* W1bT   = (__bf16*)alloc((size_t)CM * CH * 2);
  __bf16* W2aT   = (__bf16*)alloc((size_t)CM * CH * 2);
  __bf16* W2bT   = (__bf16*)alloc((size_t)CM * CH * 2);
  __bf16* WfT    = (__bf16*)alloc((size_t)80 * CM * 2);
  __bf16* U_bf   = (__bf16*)alloc((size_t)CO * CM * CH * 2);
  float*  hh     = (float*)alloc((size_t)CB * CS * CM * 4);
  float*  ht     = (float*)alloc((size_t)CB * CS * CM * 4);
  float*  prunef = (float*)alloc((size_t)CB * CK * CH * 4);
  __bf16* pruneb = (__bf16*)alloc((size_t)CB * CK * CH * 2);
  float*  phead  = (float*)alloc((size_t)CB * CK * CM * 4);
  float*  ptail  = (float*)alloc((size_t)CB * CK * CM * 4);
  float*  t_f32  = (float*)alloc((size_t)CB * CK * CO * CM * 4);
  __bf16* tT_bf  = (__bf16*)alloc((size_t)CB * CK * CO * CM * 2);
  __bf16* pair2b = (__bf16*)alloc((size_t)CB * CK * CK * CM * 2);
  int*    idx    = (int*)alloc((size_t)CB * CK * 4);
  float*  part   = (float*)alloc(256);

  auto cdiv = [](int a, int b) { return (a + b - 1) / b; };

  k_zero4<<<1, 32, 0, stream>>>(part);

  // bf16 conversions / transposed-weight builds
  {
    int n = CB * CS * CH;
    k_cvt_bf16<<<cdiv(n, 256), 256, 0, stream>>>(x, x_bf, n);
  }
  {
    int n = CM * CH;
    k_transpose_cvt<<<cdiv(n, 256), 256, 0, stream>>>(W1, W1aT, CH, CM, CM, 0,  CM);
    k_transpose_cvt<<<cdiv(n, 256), 256, 0, stream>>>(W1, W1bT, CH, CM, CM, CH, CM);
    k_transpose_cvt<<<cdiv(n, 256), 256, 0, stream>>>(W2, W2aT, CH, CM, CM, 0,  CM);
    k_transpose_cvt<<<cdiv(n, 256), 256, 0, stream>>>(W2, W2bT, CH, CM, CM, CH, CM);
  }
  {
    int n = 80 * CM;  // Wf^T padded to 80 columns (zero-filled)
    k_transpose_cvt<<<cdiv(n, 256), 256, 0, stream>>>(Wf, WfT, CM, 80, CV, 0, CV);
  }
  {
    int n = CO * CM * CH;  // U [O,M,H] is already the B^T layout for the t GEMM
    k_cvt_bf16<<<cdiv(n, 256), 256, 0, stream>>>(U, U_bf, n);
  }

  // hh = x @ W1[:H], ht = x @ W1[H:]   (768x768x768 GEMMs)
  k_gemm_bf16<<<144, 128, 0, stream>>>(x_bf, W1aT, hh, CH, CM, 48, 12, 1, 0, 0, 0);
  k_gemm_bf16<<<144, 128, 0, stream>>>(x_bf, W1bT, ht, CH, CM, 48, 12, 1, 0, 0, 0);

  // fused joint_score (writes d_out joint region)
  k_joint<<<dim3(12, CS, CB), 32, 0, stream>>>(hh, ht, b1, WfT, bfv, joint);

  // top-k pruning
  k_topk<<<CB, 256, 0, stream>>>(joint, jmask, idx);
  {
    int n = CB * CK * CH;
    k_prune<<<cdiv(n, 256), 256, 0, stream>>>(x, idx, prunef, pruneb);
  }

  // p_head / p_tail (256x768x768)
  k_gemm_bf16<<<48, 128, 0, stream>>>(pruneb, W2aT, phead, CH, CM, 16, 12, 1, 0, 0, 0);
  k_gemm_bf16<<<48, 128, 0, stream>>>(pruneb, W2bT, ptail, CH, CM, 16, 12, 1, 0, 0, 0);

  // t[(b,z),(o,i)] = pruned @ U^T  (256 x 15360, K=768)
  k_gemm_bf16<<<960, 128, 0, stream>>>(pruneb, U_bf, t_f32, CH, CO * CM, 16, 240, 1, 0, 0, 0);

  // pair2 (bf16) and t transpose (bf16)
  {
    int n = CB * CK * CK * CM;
    k_pair2<<<cdiv(n, 256), 256, 0, stream>>>(phead, ptail, b2, pair2b);
  }
  {
    int n = CB * CK * CO * CM;
    k_tT<<<cdiv(n, 256), 256, 0, stream>>>(t_f32, tT_bf);
  }

  // q_score: batched [4096 x 1280, K=768] per b, written to d_out q region
  k_gemm_bf16<<<5120, 128, 0, stream>>>(pair2b, tT_bf, qsc, CH, CK * CO, 256, 20, CB,
                                        (long long)CK * CK * CM,
                                        (long long)CK * CO * CM,
                                        (long long)CK * CK * CK * CO);

  // losses
  k_ce_joint<<<cdiv(CB * CS * CS, 256), 256, 0, stream>>>(joint, jlab, jmask, part);
  k_ce_q<<<cdiv(CB * CK * CK * CK, 256), 256, 0, stream>>>(qsc, qlab, qmask, idx, part);
  k_final<<<1, 1, 0, stream>>>(part, out);
}
